// GNN_Module_26182120636866
// MI455X (gfx1250) — compile-verified, matrix operand order checked by reading, not run on hardware
//
#include <hip/hip_runtime.h>
#include <hip/hip_bf16.h>
#include <math.h>

// ---------------------------------------------------------------------------
// GNN message-passing + GRU for MI455X (gfx1250, wave32, WMMA bf16)
// ---------------------------------------------------------------------------

typedef __attribute__((ext_vector_type(16))) __bf16 bf16x16;
typedef __attribute__((ext_vector_type(4)))  __bf16 bf16x4;
typedef __attribute__((ext_vector_type(8)))  float  f32x8;

#define D_F   128          // node feature dim
#define E_F   32           // edge feature dim
#define KIN   160          // D_F + E_F
#define MSG_F 128          // message / attention width
#define GATES 384          // 3 * D_F

__device__ __forceinline__ f32x8 wmma_bf16(bf16x16 a, bf16x16 b, f32x8 c) {
    // D = A(16x32 bf16) x B(32x16 bf16) + C(16x16 f32)
    return __builtin_amdgcn_wmma_f32_16x16x32_bf16(
        /*neg_a=*/false, a, /*neg_b=*/false, b,
        /*c_mod=*/(short)0, c, /*reuse_a=*/false, /*reuse_b=*/false);
}

__device__ __forceinline__ float sigm(float x) { return 1.0f / (1.0f + __expf(-x)); }

// ---------------------------------------------------------------------------
// Edge kernel: per 16-edge tile, build bf16 edge-input in LDS, run both
// 2-layer MLPs (msg + attention) with WMMA, sigmoid-gate, scatter-add (f32
// atomics; state_msg is L2-resident at 25.6MB << 192MB L2).
// ---------------------------------------------------------------------------
__global__ __launch_bounds__(256) void edge_msg_kernel(
    const float* __restrict__ state,
    const int*   __restrict__ src, const int* __restrict__ dst,
    const float* __restrict__ edge_feat,
    const float* __restrict__ W1, const float* __restrict__ b1,
    const float* __restrict__ W2, const float* __restrict__ b2,
    const float* __restrict__ A1, const float* __restrict__ ab1,
    const float* __restrict__ A2, const float* __restrict__ ab2,
    float* __restrict__ out_msg, int M)
{
    extern __shared__ __align__(128) char smem[];
    __bf16* W1t   = (__bf16*)(smem);            // [128][160]  40960 B
    __bf16* W2t   = (__bf16*)(smem +  40960);   // [128][128]  32768 B
    __bf16* A1t   = (__bf16*)(smem +  73728);   // [128][160]  40960 B
    __bf16* A2t   = (__bf16*)(smem + 114688);   // [128][128]  32768 B
    __bf16* Atile = (__bf16*)(smem + 147456);   // 8 * [16][160] 40960 B
    __bf16* HidM  = (__bf16*)(smem + 188416);   // 8 * [16][128] 32768 B
    __bf16* HidA  = (__bf16*)(smem + 221184);   // 8 * [16][128] 32768 B
    int*    DstI  = (int*)   (smem + 253952);   // 8 * 16 ints     512 B
    // total 254464 B dynamic LDS (one workgroup per WGP; 320KB available)

    // cooperative load: weights f32 -> bf16, transposed to [n][k] so B
    // fragments are contiguous 32B per lane
    for (int idx = threadIdx.x; idx < KIN * MSG_F; idx += blockDim.x) {
        int k = idx / MSG_F, n = idx % MSG_F;
        W1t[n * KIN + k] = (__bf16)W1[idx];
        A1t[n * KIN + k] = (__bf16)A1[idx];
    }
    for (int idx = threadIdx.x; idx < MSG_F * MSG_F; idx += blockDim.x) {
        int k = idx / MSG_F, n = idx % MSG_F;
        W2t[n * MSG_F + k] = (__bf16)W2[idx];
        A2t[n * MSG_F + k] = (__bf16)A2[idx];
    }
    __syncthreads();

    const int lane = threadIdx.x & 31;
    const int wave = threadIdx.x >> 5;
    const int r    = lane & 15;    // row (A) / col (B,C) within tile
    const int half = lane >> 4;    // K-half selector per WMMA layout

    __bf16* At   = Atile + wave * (16 * KIN);
    __bf16* Hm   = HidM  + wave * (16 * MSG_F);
    __bf16* Ha   = HidA  + wave * (16 * MSG_F);
    int*    DstW = DstI  + wave * 16;

    // tile-invariant epilogue biases: hoist to registers (per-lane column r)
    float bM1[8], bA1[8], bM2[8], bA2[8];
    #pragma unroll
    for (int t = 0; t < 8; ++t) {
        bM1[t] = b1 [t * 16 + r];
        bA1[t] = ab1[t * 16 + r];
        bM2[t] = b2 [t * 16 + r];
        bA2[t] = ab2[t * 16 + r];
    }

    const int ntiles = (M + 15) >> 4;
    for (int tile = blockIdx.x * 8 + wave; tile < ntiles; tile += gridDim.x * 8) {
        const int e0 = tile * 16;
        const int e  = e0 + r;
        const int ec = (e < M) ? e : (M - 1);
        const bool valid = (e < M);
        const int sn = src[ec], dn = dst[ec];
        if (half == 0) DstW[r] = valid ? dn : -1;

        // build edge_input = [state[src]-state[dst], edge_feat] as bf16
        // vectorized: float4 global loads, bf16x4 packed LDS stores
        const float4* srow4 = (const float4*)(state + (size_t)sn * D_F);
        const float4* drow4 = (const float4*)(state + (size_t)dn * D_F);
        #pragma unroll
        for (int c4 = 0; c4 < 16; ++c4) {
            int c = half * 16 + c4;                 // float4 column index
            float4 sv = srow4[c];
            float4 dv = drow4[c];
            bf16x4 o;
            if (valid) {
                o[0] = (__bf16)(sv.x - dv.x);
                o[1] = (__bf16)(sv.y - dv.y);
                o[2] = (__bf16)(sv.z - dv.z);
                o[3] = (__bf16)(sv.w - dv.w);
            } else {
                o[0] = (__bf16)0.0f; o[1] = (__bf16)0.0f;
                o[2] = (__bf16)0.0f; o[3] = (__bf16)0.0f;
            }
            *(bf16x4*)(At + r * KIN + c * 4) = o;
        }
        const float4* erow4 = (const float4*)(edge_feat + (size_t)ec * E_F);
        #pragma unroll
        for (int c4 = 0; c4 < 4; ++c4) {
            int c = half * 4 + c4;
            float4 v = erow4[c];
            bf16x4 o;
            if (valid) {
                o[0] = (__bf16)v.x; o[1] = (__bf16)v.y;
                o[2] = (__bf16)v.z; o[3] = (__bf16)v.w;
            } else {
                o[0] = (__bf16)0.0f; o[1] = (__bf16)0.0f;
                o[2] = (__bf16)0.0f; o[3] = (__bf16)0.0f;
            }
            *(bf16x4*)(At + r * KIN + D_F + c * 4) = o;
        }
        asm volatile("s_wait_dscnt 0" ::: "memory");

        // A fragments (K = 160 -> 5 WMMA k-steps), reused by msg AND att
        bf16x16 afr[5];
        #pragma unroll
        for (int k = 0; k < 5; ++k)
            afr[k] = *(const bf16x16*)(At + r * KIN + k * 32 + half * 16);

        // stage 1: hidden = relu(edge_in @ W1 + b1); attention branch too
        #pragma unroll
        for (int t = 0; t < 8; ++t) {
            f32x8 accM = {}; f32x8 accA = {};
            #pragma unroll
            for (int k = 0; k < 5; ++k) {
                bf16x16 bm = *(const bf16x16*)(W1t + (t * 16 + r) * KIN + k * 32 + half * 16);
                bf16x16 ba = *(const bf16x16*)(A1t + (t * 16 + r) * KIN + k * 32 + half * 16);
                accM = wmma_bf16(afr[k], bm, accM);
                accA = wmma_bf16(afr[k], ba, accA);
            }
            #pragma unroll
            for (int rr = 0; rr < 8; ++rr) {
                int m = rr + half * 8;
                float vm = accM[rr] + bM1[t]; vm = vm > 0.0f ? vm : 0.0f;
                float va = accA[rr] + bA1[t]; va = va > 0.0f ? va : 0.0f;
                Hm[m * MSG_F + t * 16 + r] = (__bf16)vm;
                Ha[m * MSG_F + t * 16 + r] = (__bf16)va;
            }
        }
        asm volatile("s_wait_dscnt 0" ::: "memory");

        // destination rows for the scatter, staged once per tile
        int dloc[8];
        #pragma unroll
        for (int rr = 0; rr < 8; ++rr) dloc[rr] = DstW[rr + half * 8];

        // stage 2: msg = hid @ W2 + b2 ; att = sigmoid(hidA @ A2 + ab2)
        bf16x16 hm[4], ha[4];
        #pragma unroll
        for (int k = 0; k < 4; ++k) {
            hm[k] = *(const bf16x16*)(Hm + r * MSG_F + k * 32 + half * 16);
            ha[k] = *(const bf16x16*)(Ha + r * MSG_F + k * 32 + half * 16);
        }
        #pragma unroll
        for (int t = 0; t < 8; ++t) {
            f32x8 am = {}; f32x8 aa = {};
            #pragma unroll
            for (int k = 0; k < 4; ++k) {
                bf16x16 bm = *(const bf16x16*)(W2t + (t * 16 + r) * MSG_F + k * 32 + half * 16);
                bf16x16 ba = *(const bf16x16*)(A2t + (t * 16 + r) * MSG_F + k * 32 + half * 16);
                am = wmma_bf16(hm[k], bm, am);
                aa = wmma_bf16(ha[k], ba, aa);
            }
            #pragma unroll
            for (int rr = 0; rr < 8; ++rr) {
                int dd = dloc[rr];
                if (dd >= 0) {
                    float msgv = am[rr] + bM2[t];
                    float attv = sigm(aa[rr] + bA2[t]);
                    atomicAdd(out_msg + (size_t)dd * MSG_F + t * 16 + r, msgv * attv);
                }
            }
        }
    }
}

// ---------------------------------------------------------------------------
// GRU kernel: gi = x @ Wih + bih, gh = h @ Whh + bhh, gates in f32.
// Six WMMA accumulators per column tile (r/z/n x ih/hh) share the A frags.
// Weights transposed bf16 in LDS: [384][128] x2 = 196608 B.
// ---------------------------------------------------------------------------
__global__ __launch_bounds__(256) void gru_kernel(
    const __bf16* __restrict__ xb,   // state_msg (bf16) [N,128]
    const __bf16* __restrict__ hb,   // state (bf16)     [N,128]
    const float* __restrict__ Wih, const float* __restrict__ bih,
    const float* __restrict__ Whh, const float* __restrict__ bhh,
    float* __restrict__ state_out, int N)
{
    extern __shared__ __align__(128) char smem[];
    __bf16* WihT = (__bf16*)(smem);           // [384][128]
    __bf16* WhhT = (__bf16*)(smem + 98304);   // [384][128]

    for (int idx = threadIdx.x; idx < D_F * GATES; idx += blockDim.x) {
        int k = idx / GATES, n = idx % GATES;
        WihT[n * D_F + k] = (__bf16)Wih[idx];
        WhhT[n * D_F + k] = (__bf16)Whh[idx];
    }
    __syncthreads();

    const int lane = threadIdx.x & 31;
    const int wave = threadIdx.x >> 5;
    const int r    = lane & 15;
    const int half = lane >> 4;

    // tile-invariant gate biases: hoist to registers (per-lane column r)
    float bIR[8], bIZ[8], bIN[8], bHR[8], bHZ[8], bHN[8];
    #pragma unroll
    for (int t = 0; t < 8; ++t) {
        int col = t * 16 + r;
        bIR[t] = bih[col]; bIZ[t] = bih[D_F + col]; bIN[t] = bih[2 * D_F + col];
        bHR[t] = bhh[col]; bHZ[t] = bhh[D_F + col]; bHN[t] = bhh[2 * D_F + col];
    }

    const int ntiles = (N + 15) >> 4;
    for (int tile = blockIdx.x * 8 + wave; tile < ntiles; tile += gridDim.x * 8) {
        int row = tile * 16 + r;
        int rc  = (row < N) ? row : (N - 1);

        bf16x16 ax[4], ah[4];
        #pragma unroll
        for (int k = 0; k < 4; ++k) {
            ax[k] = *(const bf16x16*)(xb + (size_t)rc * D_F + k * 32 + half * 16);
            ah[k] = *(const bf16x16*)(hb + (size_t)rc * D_F + k * 32 + half * 16);
        }

        #pragma unroll
        for (int t = 0; t < 8; ++t) {
            f32x8 gir = {}, giz = {}, gin = {}, ghr = {}, ghz = {}, ghn = {};
            #pragma unroll
            for (int k = 0; k < 4; ++k) {
                int nR = 0 * D_F + t * 16 + r;
                int nZ = 1 * D_F + t * 16 + r;
                int nN = 2 * D_F + t * 16 + r;
                bf16x16 bir = *(const bf16x16*)(WihT + nR * D_F + k * 32 + half * 16);
                bf16x16 biz = *(const bf16x16*)(WihT + nZ * D_F + k * 32 + half * 16);
                bf16x16 bin = *(const bf16x16*)(WihT + nN * D_F + k * 32 + half * 16);
                bf16x16 bhr = *(const bf16x16*)(WhhT + nR * D_F + k * 32 + half * 16);
                bf16x16 bhz = *(const bf16x16*)(WhhT + nZ * D_F + k * 32 + half * 16);
                bf16x16 bhn = *(const bf16x16*)(WhhT + nN * D_F + k * 32 + half * 16);
                gir = wmma_bf16(ax[k], bir, gir);
                giz = wmma_bf16(ax[k], biz, giz);
                gin = wmma_bf16(ax[k], bin, gin);
                ghr = wmma_bf16(ah[k], bhr, ghr);
                ghz = wmma_bf16(ah[k], bhz, ghz);
                ghn = wmma_bf16(ah[k], bhn, ghn);
            }
            int col = t * 16 + r;
            #pragma unroll
            for (int rr = 0; rr < 8; ++rr) {
                int m    = rr + half * 8;
                int node = tile * 16 + m;
                if (node < N) {
                    float rg = sigm(gir[rr] + bIR[t] + ghr[rr] + bHR[t]);
                    float zg = sigm(giz[rr] + bIZ[t] + ghz[rr] + bHZ[t]);
                    float ng = tanhf(gin[rr] + bIN[t] + rg * (ghn[rr] + bHN[t]));
                    float h  = (float)hb[(size_t)node * D_F + col];
                    state_out[(size_t)node * D_F + col] = (1.0f - zg) * ng + zg * h;
                }
            }
        }
    }
}

// ---------------------------------------------------------------------------
// Elementwise helpers (float4-vectorized; element counts are multiples of 4)
// ---------------------------------------------------------------------------
__global__ void relu_kernel(float4* __restrict__ x, int n4) {
    int i = blockIdx.x * blockDim.x + threadIdx.x;
    if (i < n4) {
        float4 v = x[i];
        v.x = fmaxf(v.x, 0.0f); v.y = fmaxf(v.y, 0.0f);
        v.z = fmaxf(v.z, 0.0f); v.w = fmaxf(v.w, 0.0f);
        x[i] = v;
    }
}
__global__ void zero_kernel(float4* __restrict__ x, int n4) {
    int i = blockIdx.x * blockDim.x + threadIdx.x;
    if (i < n4) x[i] = make_float4(0.0f, 0.0f, 0.0f, 0.0f);
}
__global__ void cast2_kernel(const float4* __restrict__ a, const float4* __restrict__ b,
                             bf16x4* __restrict__ ab, bf16x4* __restrict__ bb, int n4) {
    int i = blockIdx.x * blockDim.x + threadIdx.x;
    if (i < n4) {
        float4 va = a[i], vb = b[i];
        bf16x4 oa, ob;
        oa[0] = (__bf16)va.x; oa[1] = (__bf16)va.y; oa[2] = (__bf16)va.z; oa[3] = (__bf16)va.w;
        ob[0] = (__bf16)vb.x; ob[1] = (__bf16)vb.y; ob[2] = (__bf16)vb.z; ob[3] = (__bf16)vb.w;
        ab[i] = oa; bb[i] = ob;
    }
}

// ---------------------------------------------------------------------------
extern "C" void kernel_launch(void* const* d_in, const int* in_sizes, int n_in,
                              void* d_out, int out_size, void* d_ws, size_t ws_size,
                              hipStream_t stream)
{
    const float* node_feat  = (const float*)d_in[0];
    const int*   edge_index = (const int*)  d_in[1];
    const float* edge_feat  = (const float*)d_in[2];
    const float* W1  = (const float*)d_in[3];
    const float* b1  = (const float*)d_in[4];
    const float* W2  = (const float*)d_in[5];
    const float* b2  = (const float*)d_in[6];
    const float* A1  = (const float*)d_in[7];
    const float* ab1 = (const float*)d_in[8];
    const float* A2  = (const float*)d_in[9];
    const float* ab2 = (const float*)d_in[10];
    const float* Wih = (const float*)d_in[11];
    const float* bih = (const float*)d_in[12];
    const float* Whh = (const float*)d_in[13];
    const float* bhh = (const float*)d_in[14];

    const int N = in_sizes[0] / D_F;
    const int M = in_sizes[1] / 2;
    const int L = in_sizes[3] / (KIN * MSG_F);
    const int P = 2;

    // workspace layout
    char*   ws    = (char*)d_ws;
    float*  state = (float*)(ws);                                        // [N,128] f32
    float*  smsg  = (float*)(ws + (size_t)N * D_F * 4);                  // [N,128] f32
    __bf16* msgb  = (__bf16*)(ws + (size_t)N * D_F * 8);                 // [N,128] bf16
    __bf16* stb   = (__bf16*)(ws + (size_t)N * D_F * 8 + (size_t)N * D_F * 2);

    hipMemcpyAsync(state, node_feat, (size_t)N * D_F * 4,
                   hipMemcpyDeviceToDevice, stream);

    const int* src = edge_index;
    const int* dst = edge_index + M;

    const int n4    = (N * D_F) / 4;
    const int gElem = (n4 + 255) / 256;

    for (int ii = 0; ii < L; ++ii) {
        size_t w1o  = (size_t)ii * KIN * MSG_F;
        size_t b1o  = (size_t)ii * MSG_F;
        size_t w2o  = (size_t)ii * MSG_F * MSG_F;
        size_t wiho = (size_t)ii * D_F * GATES;
        size_t biho = (size_t)ii * GATES;
        for (int p = 0; p < P; ++p) {
            if (ii > 0 && p == 0)
                relu_kernel<<<gElem, 256, 0, stream>>>((float4*)state, n4);
            zero_kernel<<<gElem, 256, 0, stream>>>((float4*)smsg, n4);
            edge_msg_kernel<<<640, 256, 254464, stream>>>(
                state, src, dst, edge_feat,
                W1 + w1o, b1 + b1o, W2 + w2o, b2 + b1o,
                A1 + w1o, ab1 + b1o, A2 + w2o, ab2 + b1o,
                smsg, M);
            cast2_kernel<<<gElem, 256, 0, stream>>>(
                (const float4*)smsg, (const float4*)state,
                (bf16x4*)msgb, (bf16x4*)stb, n4);
            gru_kernel<<<400, 256, 196608, stream>>>(
                msgb, stb, Wih + wiho, bih + biho, Whh + wiho, bhh + biho,
                state, N);
        }
    }

    hipMemcpyAsync(d_out, state, (size_t)N * D_F * 4,
                   hipMemcpyDeviceToDevice, stream);
}
